// GNNEncoder_28372553957633
// MI455X (gfx1250) — matrix operand order repaired and checked by reading, not compiled, not run on hardware
//
#include <hip/hip_runtime.h>
#include <hip/hip_bf16.h>
#include <stdint.h>

// GraphSAGE 2-layer: N=50000, D=128, H=256, O=128, E=800000
#define DIM_D 128
#define DIM_H 256
#define DIM_O 128

typedef __attribute__((ext_vector_type(16))) __bf16 v16bf;
typedef __attribute__((ext_vector_type(8)))  float  v8f;

__device__ __forceinline__ unsigned short f2bf(float f) {
  unsigned int u = __float_as_uint(f);
  unsigned int r = u + 0x7FFFu + ((u >> 16) & 1u);   // round-to-nearest-even
  return (unsigned short)(r >> 16);
}
__device__ __forceinline__ float bf2f(unsigned short h) {
  return __uint_as_float(((unsigned int)h) << 16);
}

// ---------------- utility kernels ----------------

__global__ void fill_zero_f32(float* p, long long n) {
  long long i = (long long)blockIdx.x * blockDim.x + threadIdx.x;
  if (i < n) p[i] = 0.0f;
}

__global__ void cvt_f32_bf16(const float* in, unsigned short* out, long long n) {
  long long i = (long long)blockIdx.x * blockDim.x + threadIdx.x;
  if (i < n) out[i] = f2bf(in[i]);
}

// Repack row-major W[K x NOUT] (f32) into per-lane WMMA B-operand order (bf16).
// Packed ushort index tid = ((kt*NT + nt)*32 + lane)*16 + j
//   -> source element W[kt*32 + (lane>>4)*16 + j][nt*16 + (lane&15)]
__global__ void repack_w(const float* W, unsigned short* Wp, int K, int NOUT) {
  int tid = blockIdx.x * blockDim.x + threadIdx.x;
  if (tid >= K * NOUT) return;
  int j    = tid & 15;
  int lane = (tid >> 4) & 31;
  int t    = tid >> 9;
  int NT   = NOUT >> 4;
  int nt   = t % NT;
  int kt   = t / NT;
  int k = kt * 32 + (lane >> 4) * 16 + j;
  int n = nt * 16 + (lane & 15);
  Wp[tid] = f2bf(W[(size_t)k * NOUT + n]);
}

__global__ void degree_count(const long long* dst, float* cnt, int E) {
  int e = blockIdx.x * blockDim.x + threadIdx.x;
  if (e < E) unsafeAtomicAdd(&cnt[(int)dst[e]], 1.0f);
}

// Scatter-add f32 source rows (float4 chunks). cshift: log2(F/4).
__global__ void scatter_f32(const float4* __restrict__ x4,
                            const long long* __restrict__ src,
                            const long long* __restrict__ dst,
                            float* agg, long long total, int cshift, int F) {
  long long tid = (long long)blockIdx.x * blockDim.x + threadIdx.x;
  if (tid >= total) return;
  int C = 1 << cshift;
  long long e = tid >> cshift;
  int c = (int)(tid & (C - 1));
  int s = (int)src[e], d = (int)dst[e];
  float4 v = x4[(size_t)s * C + c];
  float* ap = agg + (size_t)d * F + c * 4;
  unsafeAtomicAdd(ap + 0, v.x);
  unsafeAtomicAdd(ap + 1, v.y);
  unsafeAtomicAdd(ap + 2, v.z);
  unsafeAtomicAdd(ap + 3, v.w);
}

// Scatter-add bf16 source rows (4 bf16 per chunk via uint2). cshift: log2(F/4).
__global__ void scatter_bf16(const uint2* __restrict__ h2,
                             const long long* __restrict__ src,
                             const long long* __restrict__ dst,
                             float* agg, long long total, int cshift, int F) {
  long long tid = (long long)blockIdx.x * blockDim.x + threadIdx.x;
  if (tid >= total) return;
  int C = 1 << cshift;
  long long e = tid >> cshift;
  int c = (int)(tid & (C - 1));
  int s = (int)src[e], d = (int)dst[e];
  uint2 p = h2[(size_t)s * C + c];
  float* ap = agg + (size_t)d * F + c * 4;
  unsafeAtomicAdd(ap + 0, bf2f((unsigned short)(p.x & 0xffffu)));
  unsafeAtomicAdd(ap + 1, bf2f((unsigned short)(p.x >> 16)));
  unsafeAtomicAdd(ap + 2, bf2f((unsigned short)(p.y & 0xffffu)));
  unsafeAtomicAdd(ap + 3, bf2f((unsigned short)(p.y >> 16)));
}

// out_bf[i] = bf16(agg[i] / max(cnt[i>>fshift], 1))
__global__ void normalize_bf16(const float* __restrict__ agg,
                               const float* __restrict__ cnt,
                               unsigned short* out, long long n, int fshift) {
  long long i = (long long)blockIdx.x * blockDim.x + threadIdx.x;
  if (i >= n) return;
  float c = cnt[i >> fshift];
  c = c < 1.0f ? 1.0f : c;
  out[i] = f2bf(agg[i] / c);
}

// ---------------- dual-GEMM WMMA kernel ----------------
// out = [relu]( Aa @ Wl + Ab @ Wr + bias ); Aa/Ab: [M x K] bf16 row-major.
// Wl/Wr: pre-packed B-operand bf16; bias: f32[NOUT].
// Block = 256 threads (8 waves). Block M-tile = 16*MSUB rows staged in dynamic LDS.
// Each wave owns a 16-col tile (looped if NT > 8) and MSUB independent
// accumulator chains -> back-to-back independent v_wmma issue, B frag reused
// MSUB times per load.
template <int K, int NOUT, int MSUB, bool RELU, bool OUT_BF16>
__global__ __launch_bounds__(256) void gemm_dual(
    const unsigned short* __restrict__ Aa, const unsigned short* __restrict__ Ab,
    const unsigned short* __restrict__ Wl, const unsigned short* __restrict__ Wr,
    const float* __restrict__ bias, void* outv, int M) {
  constexpr int NT   = NOUT / 16;   // column tiles
  constexpr int KT   = K / 32;      // K steps
  constexpr int QROW = K / 8;       // uint4 (8 bf16) per row
  constexpr int ROWS = 16 * MSUB;   // rows per block
  constexpr int NQ   = ROWS * QROW; // uint4 per A tile

  extern __shared__ uint4 smem[];
  uint4* sA0 = smem;        // agg-side A tile
  uint4* sA1 = smem + NQ;   // x-side A tile

  const int tid   = threadIdx.x;
  const int mBase = blockIdx.x * ROWS;

  // Stage both ROWS x K bf16 A tiles into LDS cooperatively (b128 traffic).
  for (int i = tid; i < 2 * NQ; i += 256) {
    int mat = i >= NQ;
    int rem = mat ? i - NQ : i;
    int row = rem / QROW;
    int qc  = rem - row * QROW;
    int grow = mBase + row;
    if (grow >= M) grow = M - 1;  // clamp (harmless, stores are guarded)
    const unsigned short* sp = (mat ? Ab : Aa) + (size_t)grow * K + qc * 8;
    (mat ? sA1 : sA0)[row * QROW + qc] = *(const uint4*)sp;
  }
  __syncthreads();

  const int lane = tid & 31;
  const int wave = tid >> 5;
  const int rr   = lane & 15;   // A row / output col within tile
  const int half = lane >> 4;   // K-half select

  union Frag { uint4 q[2]; v16bf v; };

  for (int nt = wave; nt < NT; nt += 8) {
    float bv = bias[nt * 16 + rr];
    v8f acc[MSUB];
#pragma unroll
    for (int s = 0; s < MSUB; ++s) acc[s] = (v8f){bv, bv, bv, bv, bv, bv, bv, bv};

#pragma unroll
    for (int kt = 0; kt < KT; ++kt) {
      // B fragments: 32 contiguous bytes per lane from packed weights (L2-hot)
      Frag wl, wr;
      const uint4* pwl = (const uint4*)Wl + ((size_t)(kt * NT + nt) * 32 + lane) * 2;
      wl.q[0] = pwl[0];
      wl.q[1] = pwl[1];
      const uint4* pwr = (const uint4*)Wr + ((size_t)(kt * NT + nt) * 32 + lane) * 2;
      wr.q[0] = pwr[0];
      wr.q[1] = pwr[1];

#pragma unroll
      for (int s = 0; s < MSUB; ++s) {
        // A fragments from LDS: 16 contiguous K at kt*32 + half*16
        Frag fa, fb;
        const uint4* pa = &sA0[(s * 16 + rr) * QROW + kt * 4 + half * 2];
        fa.q[0] = pa[0];
        fa.q[1] = pa[1];
        const uint4* pb = &sA1[(s * 16 + rr) * QROW + kt * 4 + half * 2];
        fb.q[0] = pb[0];
        fb.q[1] = pb[1];
        acc[s] = __builtin_amdgcn_wmma_f32_16x16x32_bf16(false, fa.v, false, wl.v,
                                                         (short)0, acc[s], false, false);
        acc[s] = __builtin_amdgcn_wmma_f32_16x16x32_bf16(false, fb.v, false, wr.v,
                                                         (short)0, acc[s], false, false);
      }
    }

#pragma unroll
    for (int s = 0; s < MSUB; ++s) {
#pragma unroll
      for (int v = 0; v < 8; ++v) {
        float val = acc[s][v];
        if (RELU) val = val > 0.0f ? val : 0.0f;
        int orow = mBase + s * 16 + v + half * 8;  // C/D layout: VGPR v -> M=v (+8 hi lanes)
        int ocol = nt * 16 + rr;
        if (orow < M) {
          if (OUT_BF16)
            ((unsigned short*)outv)[(size_t)orow * NOUT + ocol] = f2bf(val);
          else
            ((float*)outv)[(size_t)orow * NOUT + ocol] = val;
        }
      }
    }
  }
}

// ---------------- launcher ----------------

extern "C" void kernel_launch(void* const* d_in, const int* in_sizes, int n_in,
                              void* d_out, int out_size, void* d_ws, size_t ws_size,
                              hipStream_t stream) {
  (void)n_in; (void)out_size; (void)ws_size;

  const float*     x   = (const float*)d_in[0];
  const long long* ei  = (const long long*)d_in[1];
  const float*     W1l = (const float*)d_in[2];
  const float*     W1r = (const float*)d_in[3];
  const float*     b1  = (const float*)d_in[4];
  const float*     W2l = (const float*)d_in[5];
  const float*     W2r = (const float*)d_in[6];
  const float*     b2  = (const float*)d_in[7];

  const int N = in_sizes[0] / DIM_D;  // 50000
  const int E = in_sizes[1] / 2;      // 800000
  const long long* src = ei;
  const long long* dst = ei + E;

  // workspace carve-out
  char* w = (char*)d_ws;
  auto alloc = [&](size_t bytes) -> void* {
    void* p = (void*)w;
    w += (bytes + 255) & ~(size_t)255;
    return p;
  };
  float*          agg   = (float*)alloc((size_t)N * DIM_H * 4);
  float*          cnt   = (float*)alloc((size_t)N * 4);
  unsigned short* x_bf  = (unsigned short*)alloc((size_t)N * DIM_D * 2);
  unsigned short* a1_bf = (unsigned short*)alloc((size_t)N * DIM_D * 2);
  unsigned short* h_bf  = (unsigned short*)alloc((size_t)N * DIM_H * 2);
  unsigned short* a2_bf = (unsigned short*)alloc((size_t)N * DIM_H * 2);
  unsigned short* Wp1l  = (unsigned short*)alloc((size_t)DIM_D * DIM_H * 2);
  unsigned short* Wp1r  = (unsigned short*)alloc((size_t)DIM_D * DIM_H * 2);
  unsigned short* Wp2l  = (unsigned short*)alloc((size_t)DIM_H * DIM_O * 2);
  unsigned short* Wp2r  = (unsigned short*)alloc((size_t)DIM_H * DIM_O * 2);

  auto blocks = [](long long n) { return (unsigned int)((n + 255) / 256); };

  // x -> bf16; repack all weights
  cvt_f32_bf16<<<blocks((long long)N * DIM_D), 256, 0, stream>>>(x, x_bf, (long long)N * DIM_D);
  repack_w<<<blocks(DIM_D * DIM_H), 256, 0, stream>>>(W1l, Wp1l, DIM_D, DIM_H);
  repack_w<<<blocks(DIM_D * DIM_H), 256, 0, stream>>>(W1r, Wp1r, DIM_D, DIM_H);
  repack_w<<<blocks(DIM_H * DIM_O), 256, 0, stream>>>(W2l, Wp2l, DIM_H, DIM_O);
  repack_w<<<blocks(DIM_H * DIM_O), 256, 0, stream>>>(W2r, Wp2r, DIM_H, DIM_O);

  // degrees (shared by both layers)
  fill_zero_f32<<<blocks(N), 256, 0, stream>>>(cnt, N);
  degree_count<<<blocks(E), 256, 0, stream>>>(dst, cnt, E);

  constexpr int MSUB = 5;                  // 80-row block tile; 50000/80 = 625 exact
  const int gemmBlocks = (N + 16 * MSUB - 1) / (16 * MSUB);

  // ---- layer 1 ----
  fill_zero_f32<<<blocks((long long)N * DIM_D), 256, 0, stream>>>(agg, (long long)N * DIM_D);
  {
    long long total = (long long)E * (DIM_D / 4);
    scatter_f32<<<blocks(total), 256, 0, stream>>>((const float4*)x, src, dst, agg,
                                                   total, 5 /*log2(128/4)*/, DIM_D);
  }
  normalize_bf16<<<blocks((long long)N * DIM_D), 256, 0, stream>>>(
      agg, cnt, a1_bf, (long long)N * DIM_D, 7 /*log2(128)*/);
  {
    size_t sh = (size_t)2 * 16 * MSUB * DIM_D * 2;  // 40 KB
    gemm_dual<DIM_D, DIM_H, MSUB, true, true><<<gemmBlocks, 256, sh, stream>>>(
        a1_bf, x_bf, Wp1l, Wp1r, b1, (void*)h_bf, N);
  }

  // ---- layer 2 ----
  fill_zero_f32<<<blocks((long long)N * DIM_H), 256, 0, stream>>>(agg, (long long)N * DIM_H);
  {
    long long total = (long long)E * (DIM_H / 4);
    scatter_bf16<<<blocks(total), 256, 0, stream>>>((const uint2*)h_bf, src, dst, agg,
                                                    total, 6 /*log2(256/4)*/, DIM_H);
  }
  normalize_bf16<<<blocks((long long)N * DIM_H), 256, 0, stream>>>(
      agg, cnt, a2_bf, (long long)N * DIM_H, 8 /*log2(256)*/);
  {
    size_t sh = (size_t)2 * 16 * MSUB * DIM_H * 2;  // 80 KB (CDNA5: up to 320 KB/WG)
    gemm_dual<DIM_H, DIM_O, MSUB, false, false><<<gemmBlocks, 256, sh, stream>>>(
        a2_bf, h_bf, Wp2l, Wp2r, b2, d_out, N);
  }
}
